// NTXentLoss_7121055776856
// MI455X (gfx1250) — matrix operand order repaired and checked by reading, not compiled, not run on hardware
//
#include <hip/hip_runtime.h>
#include <stdint.h>

// ---------------------------------------------------------------------------
// NT-Xent loss, fused, for MI455X (gfx1250, wave32, WMMA).
// zn (bf16, 2MB) stays in L2; the 8192x8192 logits matrix is never formed.
// Row-max is analytically 2.0 (self-similarity / T), so single-pass exp-sum.
// Round 3: ping-pong unroll-by-2 column loop -> no buffer-rotation moves,
//          two independent WMMA/exp chains per iteration fill hazard slots.
// ---------------------------------------------------------------------------

typedef __bf16 bf16_t;
typedef __attribute__((ext_vector_type(16))) __bf16 v16bf;
typedef __attribute__((ext_vector_type(8)))  __bf16 v8bf;
typedef __attribute__((ext_vector_type(8)))  float  v8f;

#define DIM 128
#define INV_T 2.0f                     // 1/temperature, temperature = 0.5
#define TWO_LOG2E 2.885390081777927f   // 2*log2(e): exp(2c-2) = 2^(c*K - K)

static __device__ __forceinline__ uint16_t f32_to_bf16(float f) {
    union { float f; uint32_t u; } v; v.f = f;
    uint32_t r = v.u + 0x7FFFu + ((v.u >> 16) & 1u);   // round-to-nearest-even
    return (uint16_t)(r >> 16);
}

// ---------------------------------------------------------------------------
// Kernel 1: L2-normalize rows of [z_i; z_j] -> bf16 matrix zn (N x 128).
// ---------------------------------------------------------------------------
__global__ void ntxent_normalize(const float* __restrict__ zi,
                                 const float* __restrict__ zj,
                                 uint16_t* __restrict__ zn, int Brows) {
    const int wave = threadIdx.x >> 5;
    const int lane = threadIdx.x & 31;
    const int row  = blockIdx.x * (blockDim.x >> 5) + wave;
    const int N    = 2 * Brows;
    if (row >= N) return;                       // uniform per-wave
    const float* src = (row < Brows) ? (zi + (size_t)row * DIM)
                                     : (zj + (size_t)(row - Brows) * DIM);
    float4 v = ((const float4*)src)[lane];
    float ss = v.x * v.x + v.y * v.y + v.z * v.z + v.w * v.w;
#pragma unroll
    for (int off = 16; off >= 1; off >>= 1) ss += __shfl_xor(ss, off, 32);
    float inv = 1.0f / fmaxf(sqrtf(ss), 1e-8f);
    ushort4 o;
    o.x = f32_to_bf16(v.x * inv);
    o.y = f32_to_bf16(v.y * inv);
    o.z = f32_to_bf16(v.z * inv);
    o.w = f32_to_bf16(v.w * inv);
    ((ushort4*)(zn + (size_t)row * DIM))[lane] = o;
}

// ---------------------------------------------------------------------------
// Kernel 2: fused Gram + exp-sum. Each wave owns 32 rows (two 16-row A tiles,
// K=128 -> 4 bf16 WMMA fragments each, pinned in VGPRs), loops over all N/16
// column tiles, accumulating S[r] = sum_c exp(2*dot(zn_r, zn_c) - 2).
//
// VGPR layouts per CDNA5 ISA 7.12.2 (wave32):
//   A 16x32 bf16 : lane half selects K half; element e -> K = 16*half + e (same as B)
//   B 32x16 bf16 : lane half selects K half; element e -> K = 16*half + e (code's choice, not ISA table order)
//     -> B[k][n] = zn[n][k] is a CONTIGUOUS 32B row-major load (no transpose)
//   C/D 16x16 f32: elem v -> row v + 8*half, col = lane%16
//
// Column loop is ping-pong software-pipelined: while tile ct's B fragments
// feed 8 WMMAs + 16 exp-accumulates, tile ct+1's loads are in flight in the
// other register buffer. Unroll-by-2 means no rotation moves.
// ---------------------------------------------------------------------------

// Load the 4 B fragments of column tile `ct` for this lane.
static __device__ __forceinline__ void load_btile(const bf16_t* __restrict__ zn,
                                                  size_t laneColOff, int ct,
                                                  v16bf* __restrict__ b) {
    const bf16_t* pc = zn + (size_t)ct * 16 * DIM + laneColOff;
#pragma unroll
    for (int kk = 0; kk < 4; ++kk) b[kk] = *(const v16bf*)(pc + kk * 32);
}

// 8 WMMAs over one B tile + 16 exp accumulations.
static __device__ __forceinline__ void tile_math(const v16bf* __restrict__ a0,
                                                 const v16bf* __restrict__ a1,
                                                 const v16bf* __restrict__ b,
                                                 float* __restrict__ s0,
                                                 float* __restrict__ s1) {
    v8f c0 = {0.f,0.f,0.f,0.f,0.f,0.f,0.f,0.f};
    v8f c1 = {0.f,0.f,0.f,0.f,0.f,0.f,0.f,0.f};
#pragma unroll
    for (int kk = 0; kk < 4; ++kk) {
        c0 = __builtin_amdgcn_wmma_f32_16x16x32_bf16(
                 false, a0[kk], false, b[kk], (short)0, c0, false, false);
        c1 = __builtin_amdgcn_wmma_f32_16x16x32_bf16(
                 false, a1[kk], false, b[kk], (short)0, c1, false, false);
    }
#pragma unroll
    for (int v = 0; v < 8; ++v) {
        // exp(2c - 2) == 2^(c*2log2e - 2log2e): one fma + one v_exp_f32
        s0[v] += __builtin_amdgcn_exp2f(fmaf(c0[v], TWO_LOG2E, -TWO_LOG2E));
        s1[v] += __builtin_amdgcn_exp2f(fmaf(c1[v], TWO_LOG2E, -TWO_LOG2E));
    }
}

__global__ void __launch_bounds__(128)
ntxent_expsum(const uint16_t* __restrict__ znb, float* __restrict__ S, int N) {
    const bf16_t* zn = (const bf16_t*)znb;
    const int wave = threadIdx.x >> 5;
    const int lane = threadIdx.x & 31;
    const int half = lane >> 4;
    const int l16  = lane & 15;
    const int rowBase = blockIdx.x * 128 + wave * 32;

    // Persistent A fragments for row tiles [rowBase, +16) and [rowBase+16, +16).
    v16bf a0[4], a1[4];
    const bf16_t* pr0 = zn + (size_t)(rowBase + l16) * DIM;
    const bf16_t* pr1 = zn + (size_t)(rowBase + 16 + l16) * DIM;
#pragma unroll
    for (int kk = 0; kk < 4; ++kk) {
        const int o = kk * 32 + half * 16;
        v8bf lo0 = *(const v8bf*)(pr0 + o);
        v8bf hi0 = *(const v8bf*)(pr0 + o + 8);
        a0[kk] = __builtin_shufflevector(lo0, hi0, 0,1,2,3,4,5,6,7,
                                                   8,9,10,11,12,13,14,15);
        v8bf lo1 = *(const v8bf*)(pr1 + o);
        v8bf hi1 = *(const v8bf*)(pr1 + o + 8);
        a1[kk] = __builtin_shufflevector(lo1, hi1, 0,1,2,3,4,5,6,7,
                                                   8,9,10,11,12,13,14,15);
    }

    float s0[8], s1[8];
#pragma unroll
    for (int v = 0; v < 8; ++v) { s0[v] = 0.0f; s1[v] = 0.0f; }

    const int ntiles = N >> 4;                     // 512 (even)
    const size_t laneColOff = (size_t)l16 * DIM + half * 16;

    // Ping-pong pipeline: buf0 primed with tile 0.
    v16bf buf0[4], buf1[4];
    load_btile(zn, laneColOff, 0, buf0);

    for (int ct = 0; ct < ntiles; ct += 2) {
        // Phase A: load tile ct+1 into buf1, compute on buf0 (tile ct).
        load_btile(zn, laneColOff, ct + 1, buf1);
        if (ct + 8 < ntiles)   // warm WGP$ ahead (global_prefetch_b8)
            __builtin_prefetch(zn + (size_t)(ct + 8) * 16 * DIM + laneColOff, 0, 1);
        tile_math(a0, a1, buf0, s0, s1);

        // Phase B: load tile ct+2 into buf0 (wrap keeps it unconditional),
        // compute on buf1 (tile ct+1).
        int nt = ct + 2; if (nt == ntiles) nt = 0;
        load_btile(zn, laneColOff, nt, buf0);
        tile_math(a0, a1, buf1, s0, s1);
    }

    // Reduce over the 16 lanes of each half (columns of that half), write S.
#pragma unroll
    for (int v = 0; v < 8; ++v) {
        float t0 = s0[v], t1 = s1[v];
#pragma unroll
        for (int off = 8; off >= 1; off >>= 1) {
            t0 += __shfl_xor(t0, off, 32);
            t1 += __shfl_xor(t1, off, 32);
        }
        if (l16 == 0) {                      // lanes 0 (half 0) and 16 (half 1)
            S[rowBase +      half * 8 + v] = t0;
            S[rowBase + 16 + half * 8 + v] = t1;
        }
    }
}

// ---------------------------------------------------------------------------
// Kernel 3: per-row term = log(S_r) + 2 - 2*dot(zn_r, zn_pair).
// label[r] = r mod B (first half targets its own diagonal).
// ---------------------------------------------------------------------------
__global__ void ntxent_rowterm(const uint16_t* __restrict__ znb,
                               const float* __restrict__ S,
                               float* __restrict__ term, int Brows) {
    const bf16_t* zn = (const bf16_t*)znb;
    const int wave = threadIdx.x >> 5;
    const int lane = threadIdx.x & 31;
    const int row  = blockIdx.x * (blockDim.x >> 5) + wave;
    const int N    = 2 * Brows;
    if (row >= N) return;
    const int pair = (row < Brows) ? row : (row - Brows);
    const bf16_t* pa = zn + (size_t)row  * DIM;
    const bf16_t* pb = zn + (size_t)pair * DIM;
    float acc = 0.0f;
#pragma unroll
    for (int j = 0; j < 4; ++j) {
        const int idx = lane * 4 + j;
        acc += (float)pa[idx] * (float)pb[idx];
    }
#pragma unroll
    for (int off = 16; off >= 1; off >>= 1) acc += __shfl_xor(acc, off, 32);
    if (lane == 0) term[row] = logf(S[row]) + 2.0f - INV_T * acc;
}

// ---------------------------------------------------------------------------
// Kernel 4: deterministic single-block reduction -> loss = sum(term)/N.
// ---------------------------------------------------------------------------
__global__ void ntxent_reduce(const float* __restrict__ term,
                              float* __restrict__ out, int N) {
    __shared__ float sm[256];
    float acc = 0.0f;
    for (int i = threadIdx.x; i < N; i += 256) acc += term[i];
    sm[threadIdx.x] = acc;
    __syncthreads();
    for (int s = 128; s > 0; s >>= 1) {
        if ((int)threadIdx.x < s) sm[threadIdx.x] += sm[threadIdx.x + s];
        __syncthreads();
    }
    if (threadIdx.x == 0) out[0] = sm[0] / (float)N;
}

// ---------------------------------------------------------------------------
extern "C" void kernel_launch(void* const* d_in, const int* in_sizes, int n_in,
                              void* d_out, int out_size, void* d_ws, size_t ws_size,
                              hipStream_t stream) {
    const float* zi = (const float*)d_in[0];
    const float* zj = (const float*)d_in[1];
    const int Brows = in_sizes[0] / DIM;   // 4096
    const int N     = 2 * Brows;           // 8192

    // Workspace carve-up: zn bf16 (N*128*2B = 2MB), S (N f32), term (N f32).
    uint16_t* zn   = (uint16_t*)d_ws;
    float*    S    = (float*)((char*)d_ws + (size_t)N * DIM * sizeof(uint16_t));
    float*    term = S + N;

    ntxent_normalize<<<(N + 7) / 8, 256, 0, stream>>>(zi, zj, zn, Brows);
    ntxent_expsum   <<<N / 128,     128, 0, stream>>>(zn, S, N);
    ntxent_rowterm  <<<(N + 7) / 8, 256, 0, stream>>>(zn, S, term, Brows);
    ntxent_reduce   <<<1,           256, 0, stream>>>(term, (float*)d_out, N);
}
